// MHA_26826365731363
// MI455X (gfx1250) — matrix operand order repaired and checked by reading, not compiled, not run on hardware
//
#include <hip/hip_runtime.h>
#include <hip/hip_bf16.h>
#include <math.h>
#include <stdint.h>

// ---------------------------------------------------------------------------
// MHA forward for MI455X (gfx1250): bf16 WMMA (f32 accum) for all matmuls,
// flash-attention online-softmax inner loop with TDM / async-to-LDS staging,
// fp32 softmax/rmsnorm/rope.
// ---------------------------------------------------------------------------

typedef __attribute__((ext_vector_type(16))) __bf16 v16bf;
typedef __attribute__((ext_vector_type(8)))  __bf16 v8bf;
typedef __attribute__((ext_vector_type(8)))  float  v8f;
typedef unsigned int u32x4 __attribute__((ext_vector_type(4)));
typedef int          i32x8 __attribute__((ext_vector_type(8)));
typedef int          i32x4 __attribute__((ext_vector_type(4)));

#ifndef HAVE_TDM
# if defined(__has_builtin)
#  if __has_builtin(__builtin_amdgcn_tensor_load_to_lds)
#   define HAVE_TDM 1
#  else
#   define HAVE_TDM 0
#  endif
# else
#  define HAVE_TDM 0
# endif
#endif

#define SHUF16(lo, hi) __builtin_shufflevector(lo, hi, 0,1,2,3,4,5,6,7,8,9,10,11,12,13,14,15)

__device__ __forceinline__ void wave_lds_fence() {
  // Same-wave cross-lane LDS visibility: wait for this wave's DS ops.
  __builtin_amdgcn_wave_barrier();
  asm volatile("s_wait_dscnt 0" ::: "memory");
  __builtin_amdgcn_wave_barrier();
}

__device__ __forceinline__ void wait_async0() {
  asm volatile("s_wait_asynccnt 0x0" ::: "memory");
}

__device__ __forceinline__ void wait_tensor0() {
#if defined(__has_builtin)
# if __has_builtin(__builtin_amdgcn_s_wait_tensorcnt)
  __builtin_amdgcn_s_wait_tensorcnt(0);
# else
  asm volatile("s_wait_tensorcnt 0x0" ::: "memory");
# endif
#else
  asm volatile("s_wait_tensorcnt 0x0" ::: "memory");
#endif
}

// Per-lane async DMA: 16 bytes global -> LDS, tracked by ASYNCcnt.
__device__ __forceinline__ void async_g2l_b128(void* lds_generic, const void* gaddr) {
  unsigned lv = (unsigned)(uintptr_t)lds_generic;   // low 32 bits = LDS offset
  asm volatile("global_load_async_to_lds_b128 %0, %1, off" :: "v"(lv), "v"(gaddr) : "memory");
}

// ---------------------------------------------------------------------------
// fp32 -> bf16 convert
// ---------------------------------------------------------------------------
__global__ void cvt_f32_bf16(const float* __restrict__ s, __bf16* __restrict__ d, size_t n) {
  size_t i = (size_t)blockIdx.x * blockDim.x + threadIdx.x;
  size_t st = (size_t)gridDim.x * blockDim.x;
  for (; i < n; i += st) d[i] = (__bf16)s[i];
}

// ---------------------------------------------------------------------------
// C[M,N] (f32) = A[M,K](bf16) * W[N,K](bf16)^T
// Workgroup: 4 waves, 64x64 tile; wave w -> rows [tileM+16w,+16) x 64 cols.
// Fragment layouts: A 16x32 / B 32x16 bf16 in contiguous k = 16*hf + e order; C 16x16 f32.
// ---------------------------------------------------------------------------
__global__ __launch_bounds__(128) void gemm_bf16_nt(
    const __bf16* __restrict__ A, const __bf16* __restrict__ W,
    float* __restrict__ C, int M, int N, int K) {
  const int lane = threadIdx.x & 31;
  const int wave = threadIdx.x >> 5;
  const int l15  = lane & 15;
  const int hf   = lane >> 4;
  const int tileN = blockIdx.x * 64;
  const int mBase = blockIdx.y * 64 + wave * 16;

  v8f acc[4] = {};
  const __bf16* arow = A + (size_t)(mBase + l15) * K;
  for (int k0 = 0; k0 < K; k0 += 32) {
    // A fragment: lane holds row m=l15; elems -> K = k0 + hf*16 + e (contiguous)
    v8bf alo = *(const v8bf*)(arow + k0 + hf * 16);
    v8bf ahi = *(const v8bf*)(arow + k0 + hf * 16 + 8);
    v16bf a = SHUF16(alo, ahi);
#pragma unroll
    for (int t = 0; t < 4; ++t) {
      // B fragment: lane holds column n; elems -> K = k0 + hf*16 + e (contig in W row)
      const __bf16* wrow = W + (size_t)(tileN + t * 16 + l15) * K + k0 + hf * 16;
      v16bf b = *(const v16bf*)wrow;
      acc[t] = __builtin_amdgcn_wmma_f32_16x16x32_bf16(false, a, false, b,
                                                       (short)0, acc[t], false, false);
    }
  }
#pragma unroll
  for (int t = 0; t < 4; ++t)
#pragma unroll
    for (int r = 0; r < 8; ++r)
      C[(size_t)(mBase + hf * 8 + r) * N + tileN + t * 16 + l15] = acc[t][r];
}

// ---------------------------------------------------------------------------
// RMSNorm (+RoPE for Q/K) + [B,L,H,D] -> [B,H,L,D] transpose + bf16 cast.
// One wave32 per 128-element row, 4 elems/lane. mode: 0=Q(norm+rope),
// 1=K(norm+rope), 2=V(plain cast).
// ---------------------------------------------------------------------------
__global__ __launch_bounds__(128) void postproc(
    const float* __restrict__ src, const float* __restrict__ cosp,
    const float* __restrict__ sinp, const float* __restrict__ w,
    __bf16* __restrict__ dst, int mode, int Bc, int Lc, int Hc) {
  const int gw   = blockIdx.x * 4 + (threadIdx.x >> 5); // row id = (b*L+l)*H + h
  const int lane = threadIdx.x & 31;
  const int h  = gw % Hc;
  const int bl = gw / Hc;
  const int l  = bl % Lc;
  const int b  = bl / Lc;

  const float* row = src + (size_t)gw * 128;
  float t[4], out[4];
#pragma unroll
  for (int i = 0; i < 4; ++i) t[i] = row[lane * 4 + i];

  if (mode < 2) {
    float ss = t[0]*t[0] + t[1]*t[1] + t[2]*t[2] + t[3]*t[3];
#pragma unroll
    for (int off = 16; off; off >>= 1) ss += __shfl_xor(ss, off);
    float inv = rsqrtf(ss * (1.0f / 128.0f) + 1e-6f);
    float r[4];
#pragma unroll
    for (int i = 0; i < 4; ++i) r[i] = t[i] * inv * w[lane * 4 + i];
    const float* cb = cosp + ((size_t)b * Lc + l) * 64;
    const float* sb = sinp + ((size_t)b * Lc + l) * 64;
#pragma unroll
    for (int i = 0; i < 4; ++i) {
      float other = __shfl_xor(r[i], 16);     // element 64 away
      int d = lane * 4 + i;
      if (d < 64) out[i] = r[i] * cb[d] - other * sb[d];
      else        out[i] = other * sb[d - 64] + r[i] * cb[d - 64];
    }
  } else {
#pragma unroll
    for (int i = 0; i < 4; ++i) out[i] = t[i];
  }

  __bf16* drow = dst + (((size_t)(b * Hc + h) * Lc) + l) * 128;
#pragma unroll
  for (int i = 0; i < 4; ++i) drow[lane * 4 + i] = (__bf16)out[i];
}

// ---------------------------------------------------------------------------
// Flash attention. Grid: (L/64, H, B), 128 threads (4 waves).
// Wave w owns 16 query rows. K chunk (32x128 bf16 row-major) is staged by the
// Tensor Data Mover (one D# issue by wave 0, TENSORcnt); V chunk is staged by
// GLOBAL_LOAD_ASYNC_TO_LDS_B128 (ASYNCcnt) then transposed LDS->LDS for the
// B-fragment layout. O written as [B,L,H*D] bf16 (normalized by 1/li).
// ---------------------------------------------------------------------------
#define FD 128
__global__ __launch_bounds__(128) void flash_attn(
    const __bf16* __restrict__ qh, const __bf16* __restrict__ kh,
    const __bf16* __restrict__ vh, __bf16* __restrict__ Ob,
    int Bc, int Hc, int Lc) {
  const int qt   = blockIdx.x * 64;
  const int h    = blockIdx.y;
  const int b    = blockIdx.z;
  const int lane = threadIdx.x & 31;
  const int wave = threadIdx.x >> 5;
  const int l15  = lane & 15;
  const int hf   = lane >> 4;

  const size_t headOff = (size_t)(b * Hc + h) * Lc * FD;
  const __bf16* Q  = qh + headOff;
  const __bf16* Kp = kh + headOff;
  const __bf16* Vp = vh + headOff;

  __shared__ __bf16 sK[32 * FD];        // [key][d]   (TDM destination)
  __shared__ __bf16 sVr[32 * FD];       // [key][d]   (async-load destination)
  __shared__ __bf16 sVt[FD * 32];       // [d][key]   (transposed for B frags)
  __shared__ float  sS[4][16 * 32];
  __shared__ __bf16 sP[4][16 * 32];
  __shared__ float  sMi[4][16], sLi[4][16], sAl[4][16];

  const int mBase = qt + wave * 16;

  // Q fragments for this wave's 16 rows (A 16x32 layout, 4 d-chunks).
  v16bf qf[4];
  const __bf16* qrow = Q + (size_t)(mBase + l15) * FD;
#pragma unroll
  for (int f = 0; f < 4; ++f) {
    v8bf lo = *(const v8bf*)(qrow + f * 32 + hf * 16);
    v8bf hi = *(const v8bf*)(qrow + f * 32 + hf * 16 + 8);
    qf[f] = SHUF16(lo, hi);
  }

  v8f acc[8] = {};
  if (lane < 16) { sMi[wave][lane] = -__builtin_inff(); sLi[wave][lane] = 0.f; }
  const float scale = 0.08838834764831845f; // 1/sqrt(128)
  const int kend = qt + 64;                 // causal: keys < query-tile end

  for (int kc = 0; kc < kend; kc += 32) {
    __syncthreads();
    const __bf16* gK = Kp + (size_t)kc * FD;
    const __bf16* gV = Vp + (size_t)kc * FD;
    if (kc + 32 < kend)
      __builtin_prefetch(Kp + (size_t)(kc + 32) * FD + threadIdx.x * 32, 0, 1);

    // ---- K chunk: Tensor Data Mover DMA (32 rows x 128 cols, bf16) ----
#if HAVE_TDM
    if (wave == 0) {
      unsigned ldsK = (unsigned)(uintptr_t)(&sK[0]);
      unsigned long long ga = (unsigned long long)(uintptr_t)gK;
      u32x4 g0 = {};
      i32x8 g1 = {};
      i32x4 g2 = {0, 0, 0, 0}, g3 = {0, 0, 0, 0};
      g0[0] = 1u;                                        // count=1, no gather
      g0[1] = ldsK;                                      // lds_addr
      g0[2] = (unsigned)(ga & 0xFFFFFFFFu);              // global_addr[31:0]
      g0[3] = (unsigned)((ga >> 32) & 0x1FFFFFFu) | (2u << 30); // [56:32] | type=2
      g1[0] = 1 << 16;                                   // data_size = 2 bytes
      g1[1] = (FD & 0xFFFF) << 16;                       // tensor_dim0 = 128 (lo)
      g1[2] = (32 & 0xFFFF) << 16;                       // tensor_dim1 = 32 (lo)
      g1[3] = FD << 16;                                  // tile_dim0 = 128
      g1[4] = 32;                                        // tile_dim1 = 32, tile_dim2 = 0
      g1[5] = FD;                                        // tensor_dim0_stride = 128
      g1[6] = 0; g1[7] = 0;
# if __clang_major__ >= 23
      i32x8 g4 = {0, 0, 0, 0, 0, 0, 0, 0};
      __builtin_amdgcn_tensor_load_to_lds(g0, g1, g2, g3, g4, 0);
# else
      __builtin_amdgcn_tensor_load_to_lds(g0, g1, g2, g3, 0);
# endif
    }
#else
    {
      int tid = threadIdx.x;
#pragma unroll
      for (int i = 0; i < 4; ++i) {
        int vec = tid + i * 128, row = vec >> 4, col8 = (vec & 15) * 8;
        async_g2l_b128(&sK[row * FD + col8], gK + (size_t)row * FD + col8);
      }
    }
#endif

    // ---- V chunk: async global->LDS (rows as-is), 512 x b128 per workgroup ----
    {
      int tid = threadIdx.x;
#pragma unroll
      for (int i = 0; i < 4; ++i) {
        int vec = tid + i * 128, row = vec >> 4, col8 = (vec & 15) * 8;
        async_g2l_b128(&sVr[row * FD + col8], gV + (size_t)row * FD + col8);
      }
    }
    wait_async0();
#if HAVE_TDM
    wait_tensor0();
#endif
    __syncthreads();

    // ---- transpose V in LDS: sVr[key][d] -> sVt[d][key] ----
    {
      int tid = threadIdx.x;
#pragma unroll
      for (int i = 0; i < 4; ++i) {
        int vec = tid + i * 128, row = vec >> 4, col8 = (vec & 15) * 8;
        v8bf dv = *(const v8bf*)(&sVr[row * FD + col8]);
#pragma unroll
        for (int e = 0; e < 8; ++e) sVt[(col8 + e) * 32 + row] = dv[e];
      }
    }
    __syncthreads();

    // ---- S = Q * K^T (16x32 scores per wave) ----
#pragma unroll
    for (int nh = 0; nh < 2; ++nh) {
      v8f s = {};
#pragma unroll
      for (int f = 0; f < 4; ++f) {
        v16bf bfrag = *(const v16bf*)(&sK[(nh * 16 + l15) * FD + f * 32 + hf * 16]);
        s = __builtin_amdgcn_wmma_f32_16x16x32_bf16(false, qf[f], false, bfrag,
                                                    (short)0, s, false, false);
      }
#pragma unroll
      for (int r = 0; r < 8; ++r) {
        int ml = hf * 8 + r, nl = nh * 16 + l15;
        int ng = kc + nl, mg = mBase + ml;
        sS[wave][ml * 32 + nl] = (ng <= mg) ? s[r] * scale : -__builtin_inff();
      }
    }
    wave_lds_fence();

    // ---- online softmax: lanes 0..15, one row each ----
    if (lane < 16) {
      int m = lane;
      float mi = sMi[wave][m], li = sLi[wave][m];
      float sv[32], rmax = -__builtin_inff();
#pragma unroll
      for (int j = 0; j < 32; ++j) { sv[j] = sS[wave][m * 32 + j]; rmax = fmaxf(rmax, sv[j]); }
      float nm = fmaxf(mi, rmax);
      float alpha, rs = 0.f;
      if (nm == -__builtin_inff()) {
        alpha = 1.f;
#pragma unroll
        for (int j = 0; j < 32; ++j) sP[wave][m * 32 + j] = (__bf16)0.f;
      } else {
        alpha = __expf(mi - nm);
#pragma unroll
        for (int j = 0; j < 32; ++j) {
          float p = __expf(sv[j] - nm);
          rs += p;
          sP[wave][m * 32 + j] = (__bf16)p;
        }
      }
      sMi[wave][m] = nm;
      sLi[wave][m] = li * alpha + rs;
      sAl[wave][m] = alpha;
    }
    wave_lds_fence();

    // ---- rescale running O ----
#pragma unroll
    for (int r = 0; r < 8; ++r) {
      float a = sAl[wave][hf * 8 + r];
#pragma unroll
      for (int sl = 0; sl < 8; ++sl) acc[sl][r] *= a;
    }

    // ---- O += P * V ----
    v8bf plo = *(const v8bf*)(&sP[wave][l15 * 32 + hf * 16]);
    v8bf phi = *(const v8bf*)(&sP[wave][l15 * 32 + hf * 16 + 8]);
    v16bf pf = SHUF16(plo, phi);
#pragma unroll
    for (int sl = 0; sl < 8; ++sl) {
      v16bf vb = *(const v16bf*)(&sVt[(sl * 16 + l15) * 32 + hf * 16]);
      acc[sl] = __builtin_amdgcn_wmma_f32_16x16x32_bf16(false, pf, false, vb,
                                                        (short)0, acc[sl], false, false);
    }
  }
  wave_lds_fence();

  // ---- epilogue: normalize, store [B,L,H*D] bf16 ----
#pragma unroll
  for (int r = 0; r < 8; ++r) {
    int ml = hf * 8 + r;
    float inv = 1.0f / sLi[wave][ml];
    size_t base = (((size_t)b * Lc + (mBase + ml)) * Hc + h) * FD + l15;
#pragma unroll
    for (int sl = 0; sl < 8; ++sl) Ob[base + sl * 16] = (__bf16)(acc[sl][r] * inv);
  }
}

// ---------------------------------------------------------------------------
// host-side launch
// ---------------------------------------------------------------------------
extern "C" void kernel_launch(void* const* d_in, const int* in_sizes, int n_in,
                              void* d_out, int out_size, void* d_ws, size_t ws_size,
                              hipStream_t stream) {
  (void)in_sizes; (void)n_in; (void)out_size; (void)ws_size;
  constexpr int Bc = 4, Lc = 2048, Ec = 2048, Hc = 16, Dc = 128;
  constexpr size_t NBLE = (size_t)Bc * Lc * Ec;  // 16,777,216 (also B*L*H*D)
  constexpr size_t NEE  = (size_t)Ec * Ec;       // 4,194,304

  const float* x  = (const float*)d_in[0];
  // d_in[1] = attention_mask (causal; computed analytically, unused)
  const float* cosp = (const float*)d_in[2];
  const float* sinp = (const float*)d_in[3];
  const float* Wq = (const float*)d_in[4];
  const float* Wk = (const float*)d_in[5];
  const float* Wv = (const float*)d_in[6];
  const float* Wo = (const float*)d_in[7];
  const float* qn = (const float*)d_in[8];
  const float* kn = (const float*)d_in[9];

  char* ws = (char*)d_ws;
  size_t off = 0;
  auto take = [&](size_t bytes) { char* p = ws + off; off = (off + bytes + 255) & ~(size_t)255; return p; };
  __bf16* xb  = (__bf16*)take(NBLE * 2);
  __bf16* Wqb = (__bf16*)take(NEE * 2);
  __bf16* Wkb = (__bf16*)take(NEE * 2);
  __bf16* Wvb = (__bf16*)take(NEE * 2);
  __bf16* Wob = (__bf16*)take(NEE * 2);
  float*  Qf  = (float*) take(NBLE * 4);
  float*  Kf  = (float*) take(NBLE * 4);
  float*  Vf  = (float*) take(NBLE * 4);
  __bf16* qhb = (__bf16*)take(NBLE * 2);
  __bf16* khb = (__bf16*)take(NBLE * 2);
  __bf16* vhb = (__bf16*)take(NBLE * 2);
  __bf16* Obf = (__bf16*)take(NBLE * 2);

  // 1) fp32 -> bf16
  cvt_f32_bf16<<<2048, 256, 0, stream>>>(x,  xb,  NBLE);
  cvt_f32_bf16<<<1024, 256, 0, stream>>>(Wq, Wqb, NEE);
  cvt_f32_bf16<<<1024, 256, 0, stream>>>(Wk, Wkb, NEE);
  cvt_f32_bf16<<<1024, 256, 0, stream>>>(Wv, Wvb, NEE);
  cvt_f32_bf16<<<1024, 256, 0, stream>>>(Wo, Wob, NEE);

  // 2) projections: [B*L, E] x [H*D, E]^T
  const int M = Bc * Lc, N = Hc * Dc, K = Ec;
  dim3 ggrid(N / 64, M / 64);
  gemm_bf16_nt<<<ggrid, 128, 0, stream>>>(xb, Wqb, Qf, M, N, K);
  gemm_bf16_nt<<<ggrid, 128, 0, stream>>>(xb, Wkb, Kf, M, N, K);
  gemm_bf16_nt<<<ggrid, 128, 0, stream>>>(xb, Wvb, Vf, M, N, K);

  // 3) rmsnorm + rope + head transpose -> bf16 [B,H,L,D]
  const int rows = Bc * Lc * Hc;            // 131072, 4 waves/block
  postproc<<<rows / 4, 128, 0, stream>>>(Qf, cosp, sinp, qn, qhb, 0, Bc, Lc, Hc);
  postproc<<<rows / 4, 128, 0, stream>>>(Kf, cosp, sinp, kn, khb, 1, Bc, Lc, Hc);
  postproc<<<rows / 4, 128, 0, stream>>>(Vf, cosp, sinp, qn, vhb, 2, Bc, Lc, Hc);

  // 4) flash attention -> [B,L,H*D] bf16
  dim3 fgrid(Lc / 64, Hc, Bc);
  flash_attn<<<fgrid, 128, 0, stream>>>(qhb, khb, vhb, Obf, Bc, Hc, Lc);

  // 5) output projection: [B*L, H*D] x [E, H*D]^T -> d_out (f32)
  dim3 ogrid(Ec / 64, M / 64);
  gemm_bf16_nt<<<ogrid, 128, 0, stream>>>(Obf, Wob, (float*)d_out, M, Ec, Hc * Dc);
}
